// Attention_19292993093695
// MI455X (gfx1250) — compile-verified, matrix operand order checked by reading, not run on hardware
//
#include <hip/hip_runtime.h>

// ---------------- problem constants ----------------
constexpr int B_  = 4;
constexpr int S_  = 2048;
constexpr int D_  = 1024;
constexpr int H_  = 16;
constexpr int E_  = 1024;
constexpr int HD_ = 64;            // head dim
constexpr int N3_ = 3 * E_;        // fused QKV output dim
constexpr float SCALING_ = 0.125f; // 64^-0.5

typedef __attribute__((ext_vector_type(16))) __bf16 v16bf;
typedef __attribute__((ext_vector_type(8)))  __bf16 v8bf;
typedef __attribute__((ext_vector_type(4)))  __bf16 v4bf;
typedef __attribute__((ext_vector_type(8)))  float  v8f;
typedef __attribute__((ext_vector_type(4)))  unsigned int vu4;
typedef __attribute__((ext_vector_type(8)))  int vi8;
typedef __attribute__((ext_vector_type(4)))  int vi4;

// TDM-padded LDS row stride: 64 bf16 (128B) + 16B pad = 72 elements.
constexpr int BROW_ = 72;

// Load one WMMA A/B operand row: lane holds two contiguous 8-element bf16
// chunks at k-offsets kc and kc+16 (caller folds the +8 hi-half-wave offset
// into kc). Lowers to b128 loads (global or ds depending on address space).
static __device__ __forceinline__ v16bf load_row16(const __bf16* __restrict__ p, int kc) {
  v8bf lo = *reinterpret_cast<const v8bf*>(p + kc);
  v8bf hi = *reinterpret_cast<const v8bf*>(p + kc + 16);
  return __builtin_shufflevector(lo, hi, 0, 1, 2, 3, 4, 5, 6, 7,
                                         8, 9, 10, 11, 12, 13, 14, 15);
}

template <bool RA>
static __device__ __forceinline__ v8f wmma_bf16(v16bf a, v16bf b, v8f c) {
  return __builtin_amdgcn_wmma_f32_16x16x32_bf16(
      /*neg_a=*/false, a, /*neg_b=*/false, b,
      /*c_mod=*/(short)0, c, /*reuse_a=*/RA, /*reuse_b=*/false);
}

// ---- Tensor Data Mover: 2D tile (64 rows x 64 bf16) global -> LDS ---------
// Pads 4 DWORDs (16B) after every 32 DWORDs (128B = one row), so the LDS
// destination has a 144B row stride (bank-conflict-friendly for ds_load_b128).
// D# packing per CDNA5 ISA sect. 8. 6-arg builtin form
// (amdgpu-toolchain clang-23): (v4u, v8i, v4i, v4i, v8i, i32 cpol).
static __device__ __forceinline__ void tdm_load_b_tile(
    unsigned lds_byte_off, const __bf16* gptr, int tensor_rows, int row_stride_elems) {
  unsigned long long ga = (unsigned long long)(uintptr_t)gptr;
  vu4 g0;
  g0.x = 1u;                                   // count=1 valid descriptor
  g0.y = lds_byte_off;                         // D#.lds_addr (bytes)
  g0.z = (unsigned)(ga & 0xffffffffu);         // global_addr[31:0]
  g0.w = (unsigned)((ga >> 32) & 0x01ffffffu)  // global_addr[56:32]
         | (2u << 30);                         // type=2 ("image")
  vi8 g1;
  g1[0] = (1 << 16)                            // data_size=1 (2 bytes)
        | (1 << 20)                            // pad_enable
        | (4 << 22)                            // pad_interval: 32 DWORDs
        | (3 << 25);                           // pad_amount: 4 DWORDs (16B)
  g1[1] = (64 & 0xffff) << 16;                 // tensor_dim0[15:0]=64 (tile-local)
  g1[2] = 0 | ((tensor_rows & 0xffff) << 16);  // dim0[31:16]=0, tensor_dim1[15:0]
  g1[3] = (int)(((unsigned)tensor_rows >> 16) & 0xffffu)
        | (64 << 16);                          // tile_dim0 = 64 elements
  g1[4] = 64;                                  // tile_dim1 = 64 rows (tile_dim2=0)
  g1[5] = row_stride_elems;                    // tensor_dim0_stride[31:0]
  g1[6] = 0;                                   // stride hi, dim1_stride lo
  g1[7] = 0;
  vi4 z4 = {0, 0, 0, 0};
  vi8 z8 = {0, 0, 0, 0, 0, 0, 0, 0};
  __builtin_amdgcn_tensor_load_to_lds(g0, g1, z4, z4, z8, 0);
}

// ---------------- kernel 1: f32 -> bf16 conversion (vectorized) ------------
__global__ void cvt_f32_bf16_v4(const float4* __restrict__ src,
                                v4bf* __restrict__ dst, int n4) {
  int i = blockIdx.x * blockDim.x + threadIdx.x;
  if (i < n4) {
    float4 f = src[i];
    v4bf o;
    o[0] = (__bf16)f.x; o[1] = (__bf16)f.y;
    o[2] = (__bf16)f.z; o[3] = (__bf16)f.w;
    dst[i] = o;
  }
}

// Shared inner step: A from global row, 4 B tiles from LDS strip, 4 WMMAs
// with reuse-A hints. Scheduling groups pin the order {2 VMEM rd}{8 DS rd}
// {4 WMMA} so the B operands get distinct live ranges and the WMMA chain
// issues behind a single partial dscnt wait instead of four full drains.
static __device__ __forceinline__ void gemm_step(
    const __bf16* __restrict__ arow, int ka,
    const __bf16* __restrict__ bbuf, int kb, int col, v8f acc[4]) {
  v16bf a = load_row16(arow, ka);
  v16bf bm[4];
#pragma unroll
  for (int j = 0; j < 4; ++j)
    bm[j] = load_row16(bbuf + (j * 16 + col) * BROW_, kb);
  acc[0] = wmma_bf16<false>(a, bm[0], acc[0]);
#pragma unroll
  for (int j = 1; j < 4; ++j)
    acc[j] = wmma_bf16<true>(a, bm[j], acc[j]);
#if __has_builtin(__builtin_amdgcn_sched_group_barrier)
  __builtin_amdgcn_sched_group_barrier(0x020, 2, 0); // 2x VMEM read (A)
  __builtin_amdgcn_sched_group_barrier(0x100, 8, 0); // 8x DS read   (B)
  __builtin_amdgcn_sched_group_barrier(0x008, 4, 0); // 4x WMMA
#endif
}

// ---------------- kernel 2: fused QKV projection ----------------
// C[m, c] = sum_k x[m,k] * w_in[c,k] + b_in[c]
// Block = 128(m) x 64(c): 8 waves each own 16 m-rows; all share one 64-row
// B strip staged by the TDM into double-buffered LDS (traffic /8).
__global__ __launch_bounds__(256) void qkv_gemm(
    const __bf16* __restrict__ xb, const __bf16* __restrict__ wb,
    const float* __restrict__ b_in,
    __bf16* __restrict__ qb, __bf16* __restrict__ kb, __bf16* __restrict__ vtb) {
  __shared__ __bf16 bsm[2][64 * BROW_];        // 2 x 9 KB

  const int wave = threadIdx.x >> 5, lane = threadIdx.x & 31;
  const int col  = lane & 15;
  const int khi  = (lane >> 4) << 3;           // 0 or 8
  const int mb = blockIdx.x / (N3_ / 64);      // 64 m-blocks
  const int nb = blockIdx.x % (N3_ / 64);      // 48 n-blocks
  const int m0 = mb * 128 + wave * 16;
  const int n0 = nb * 64;

  const __bf16* arow = xb + (size_t)(m0 + col) * D_;
  const __bf16* gB   = wb + (size_t)n0 * D_;

  v8f acc[4] = {};
  constexpr int NTK = D_ / 64;                 // 16 k-tiles
  if (wave == 0)
    tdm_load_b_tile((unsigned)(uintptr_t)&bsm[0][0], gB, N3_ - n0, D_);

  for (int t = 0; t < NTK; ++t) {
    if (wave == 0) __builtin_amdgcn_s_wait_tensorcnt(0);
    __syncthreads();                           // buf[t&1] ready for all waves
    if (wave == 0 && t + 1 < NTK)
      tdm_load_b_tile((unsigned)(uintptr_t)&bsm[(t + 1) & 1][0],
                      gB + (t + 1) * 64, N3_ - n0, D_);
    const __bf16* bbuf = &bsm[t & 1][0];
    const int kg = t * 64;
    gemm_step(arow, kg + khi,      bbuf, khi,      col, acc);
    gemm_step(arow, kg + 32 + khi, bbuf, 32 + khi, col, acc);
    __syncthreads();                           // done reading buf[t&1]
  }

  const int rbase = (lane >> 4) << 3;
#pragma unroll
  for (int j = 0; j < 4; ++j) {
    const int c = n0 + j * 16 + col;
    const float bias = b_in[c];
#pragma unroll
    for (int r = 0; r < 8; ++r) {
      const int m = m0 + rbase + r;
      const int s = m & (S_ - 1);
      const int bb = m >> 11;                  // m / S_
      const float v = acc[j][r] + bias;
      if (c < E_) {
        const int h = c >> 6, d = c & 63;
        qb[(((size_t)bb * H_ + h) * S_ + s) * HD_ + d] = (__bf16)(v * SCALING_);
      } else if (c < 2 * E_) {
        const int e = c - E_, h = e >> 6, d = e & 63;
        kb[(((size_t)bb * H_ + h) * S_ + s) * HD_ + d] = (__bf16)v;
      } else {
        const int e = c - 2 * E_, h = e >> 6, d = e & 63;
        vtb[(((size_t)bb * H_ + h) * HD_ + d) * S_ + s] = (__bf16)v; // V^T
      }
    }
  }
}

// ---------------- kernel 3: fused scores + softmax + weights + P@V ----------
// One block per (b, h, 16 q-rows). 8 waves. 128 KB static LDS holds the
// full 16 x 2048 probability strip (reused for the cross-wave PV reduction).
__global__ __launch_bounds__(256) void attn_kernel(
    const __bf16* __restrict__ qb, const __bf16* __restrict__ kb,
    const __bf16* __restrict__ vtb, float* __restrict__ wout,
    __bf16* __restrict__ ab) {
  __shared__ float lds[16 * S_]; // 128 KB

  const int tid  = threadIdx.x;
  const int wave = tid >> 5, lane = tid & 31;
  const int col  = lane & 15;
  const int khi  = (lane >> 4) << 3;
  const int qt   = blockIdx.x & 127;     // S/16 tiles
  const int bh   = blockIdx.x >> 7;
  const int b    = bh >> 4, h = bh & 15;
  const int q0   = qt << 4;

  // ---- scores: S[16 x 2048] = Q_tile (pre-scaled) @ K^T ----
  const __bf16* qrow = qb + ((size_t)bh * S_ + (q0 + col)) * HD_;
  const v16bf a0 = load_row16(qrow, khi);
  const v16bf a1 = load_row16(qrow, 32 + khi);
  const int rbase = (lane >> 4) << 3;
  for (int ct = wave; ct < S_ / 16; ct += 8) {
    const int k0 = ct << 4;
    const __bf16* krow = kb + ((size_t)bh * S_ + (k0 + col)) * HD_;
    v16bf b0 = load_row16(krow, khi);
    v16bf b1 = load_row16(krow, 32 + khi);
    v8f c = {};
    c = wmma_bf16<false>(a0, b0, c);
    c = wmma_bf16<false>(a1, b1, c);
#pragma unroll
    for (int r = 0; r < 8; ++r)
      lds[(rbase + r) * S_ + k0 + col] = c[r];
  }
  __syncthreads();

  // ---- row softmax: 16 threads per row, shfl_xor reductions (wave32) ----
  {
    const int row = tid >> 4, sub = tid & 15;
    float* p = lds + row * S_;
    float mx = -3.4e38f;
    for (int c = sub; c < S_; c += 16) mx = fmaxf(mx, p[c]);
#pragma unroll
    for (int off = 8; off >= 1; off >>= 1) mx = fmaxf(mx, __shfl_xor(mx, off, 32));
    float sum = 0.f;
    for (int c = sub; c < S_; c += 16) {
      float e = __expf(p[c] - mx);
      p[c] = e;
      sum += e;
    }
#pragma unroll
    for (int off = 8; off >= 1; off >>= 1) sum += __shfl_xor(sum, off, 32);
    const float inv = 1.0f / sum;
    for (int c = sub; c < S_; c += 16) p[c] *= inv;
  }
  __syncthreads();

  // ---- stream attention weights to global (coalesced float4) ----
  {
    float4* dst = reinterpret_cast<float4*>(wout + ((size_t)bh * S_ + q0) * S_);
    const float4* src = reinterpret_cast<const float4*>(lds);
    for (int i = tid; i < (16 * S_) / 4; i += 256) dst[i] = src[i];
  }

  // ---- P @ V: each wave owns 256 of the 2048 K-dim, 4 n-tiles of V^T ----
  v8f acc[4] = {};
  const float* prow = lds + col * S_;    // A-matrix row = lane & 15
  const __bf16* vrow0 = vtb + ((size_t)bh * HD_ + col) * S_;
  const int kstart = wave * 256;
  for (int kk = kstart; kk < kstart + 256; kk += 32) {
    const int kc = kk + khi;
    v16bf a;
#pragma unroll
    for (int i = 0; i < 8; ++i) {
      a[i]     = (__bf16)prow[kc + i];
      a[i + 8] = (__bf16)prow[kc + 16 + i];
    }
    v16bf bm[4];
#pragma unroll
    for (int j = 0; j < 4; ++j)
      bm[j] = load_row16(vrow0 + (size_t)(j * 16) * S_, kc);
    acc[0] = wmma_bf16<false>(a, bm[0], acc[0]);
#pragma unroll
    for (int j = 1; j < 4; ++j)
      acc[j] = wmma_bf16<true>(a, bm[j], acc[j]);
  }
  __syncthreads();           // done reading probs; reuse LDS for partials

  {
#pragma unroll
    for (int j = 0; j < 4; ++j)
#pragma unroll
      for (int r = 0; r < 8; ++r)
        lds[wave * 1024 + (rbase + r) * 64 + j * 16 + col] = acc[j][r];
  }
  __syncthreads();

  for (int i = tid; i < 1024; i += 256) {
    float s = 0.f;
#pragma unroll
    for (int w = 0; w < 8; ++w) s += lds[w * 1024 + i];
    const int m = i >> 6, d = i & 63;
    ab[((size_t)b * S_ + q0 + m) * E_ + h * HD_ + d] = (__bf16)s;
  }
}

// ---------------- kernel 4: output projection ----------------
// out[m, n] = sum_e ab[m,e] * w_out[n,e] + b_out[n]
// Same TDM-staged shared-B structure as qkv_gemm.
__global__ __launch_bounds__(256) void out_gemm(
    const __bf16* __restrict__ ab, const __bf16* __restrict__ wb,
    const float* __restrict__ b_out, float* __restrict__ out) {
  __shared__ __bf16 bsm[2][64 * BROW_];

  const int wave = threadIdx.x >> 5, lane = threadIdx.x & 31;
  const int col  = lane & 15;
  const int khi  = (lane >> 4) << 3;
  const int mb = blockIdx.x / (D_ / 64);       // 64 m-blocks
  const int nb = blockIdx.x % (D_ / 64);       // 16 n-blocks
  const int m0 = mb * 128 + wave * 16;
  const int n0 = nb * 64;

  const __bf16* arow = ab + (size_t)(m0 + col) * E_;
  const __bf16* gB   = wb + (size_t)n0 * E_;

  v8f acc[4] = {};
  constexpr int NTK = E_ / 64;
  if (wave == 0)
    tdm_load_b_tile((unsigned)(uintptr_t)&bsm[0][0], gB, D_ - n0, E_);

  for (int t = 0; t < NTK; ++t) {
    if (wave == 0) __builtin_amdgcn_s_wait_tensorcnt(0);
    __syncthreads();
    if (wave == 0 && t + 1 < NTK)
      tdm_load_b_tile((unsigned)(uintptr_t)&bsm[(t + 1) & 1][0],
                      gB + (t + 1) * 64, D_ - n0, E_);
    const __bf16* bbuf = &bsm[t & 1][0];
    const int kg = t * 64;
    gemm_step(arow, kg + khi,      bbuf, khi,      col, acc);
    gemm_step(arow, kg + 32 + khi, bbuf, 32 + khi, col, acc);
    __syncthreads();
  }

  const int rbase = (lane >> 4) << 3;
#pragma unroll
  for (int j = 0; j < 4; ++j) {
    const int c = n0 + j * 16 + col;
    const float bias = b_out[c];
#pragma unroll
    for (int r = 0; r < 8; ++r)
      out[(size_t)(m0 + rbase + r) * D_ + c] = acc[j][r] + bias;
  }
}

// ---------------- host launch ----------------
extern "C" void kernel_launch(void* const* d_in, const int* in_sizes, int n_in,
                              void* d_out, int out_size, void* d_ws, size_t ws_size,
                              hipStream_t stream) {
  const float* x     = (const float*)d_in[0];
  const float* w_in  = (const float*)d_in[1];
  const float* b_in  = (const float*)d_in[2];
  const float* w_out = (const float*)d_in[3];
  const float* b_out = (const float*)d_in[4];

  float* out     = (float*)d_out;                       // [B,S,D]
  float* weights = out + (size_t)B_ * S_ * D_;          // [B,H,S,S]

  // workspace carve-up (bf16 buffers), ~91 MB total
  char* w = (char*)d_ws;
  __bf16* xb      = (__bf16*)w; w += (size_t)B_ * S_ * D_ * 2;
  __bf16* w_in_b  = (__bf16*)w; w += (size_t)N3_ * D_ * 2;
  __bf16* w_out_b = (__bf16*)w; w += (size_t)D_ * E_ * 2;
  __bf16* qb      = (__bf16*)w; w += (size_t)B_ * H_ * S_ * HD_ * 2;
  __bf16* kb      = (__bf16*)w; w += (size_t)B_ * H_ * S_ * HD_ * 2;
  __bf16* vtb     = (__bf16*)w; w += (size_t)B_ * H_ * S_ * HD_ * 2;
  __bf16* ab      = (__bf16*)w; w += (size_t)B_ * S_ * E_ * 2;

  const int nx4  = (B_ * S_ * D_) / 4;
  const int nwi4 = (N3_ * D_) / 4;
  const int nwo4 = (D_ * E_) / 4;
  cvt_f32_bf16_v4<<<(nx4 + 255) / 256, 256, 0, stream>>>(
      (const float4*)x, (v4bf*)xb, nx4);
  cvt_f32_bf16_v4<<<(nwi4 + 255) / 256, 256, 0, stream>>>(
      (const float4*)w_in, (v4bf*)w_in_b, nwi4);
  cvt_f32_bf16_v4<<<(nwo4 + 255) / 256, 256, 0, stream>>>(
      (const float4*)w_out, (v4bf*)w_out_b, nwo4);

  // (B*S/128) m-blocks x (3E/64) n-blocks = 64*48 = 3072
  qkv_gemm<<<3072, 256, 0, stream>>>(xb, w_in_b, b_in, qb, kb, vtb);

  // B*H*(S/16) = 8192 blocks
  attn_kernel<<<B_ * H_ * (S_ / 16), 256, 0, stream>>>(qb, kb, vtb, weights, ab);

  // (B*S/128) x (D/64) = 64*16 = 1024
  out_gemm<<<1024, 256, 0, stream>>>(ab, w_out_b, b_out, out);
}
